// JointAttention_635655160357
// MI455X (gfx1250) — matrix operand order repaired and checked by reading, not compiled, not run on hardware
//
#include <hip/hip_runtime.h>

typedef __attribute__((ext_vector_type(16))) _Float16 v16h;
typedef __attribute__((ext_vector_type(8)))  _Float16 v8h;
typedef __attribute__((ext_vector_type(8)))  float    v8f;
typedef __attribute__((ext_vector_type(4)))  unsigned u32x4;
typedef __attribute__((ext_vector_type(4)))  int      i32x4;
typedef __attribute__((ext_vector_type(8)))  int      i32x8;

#define HEADS  16
#define DH     64
#define SN1    3072
#define SN2    1024
#define NTOT   4096
#define DIN1   1024
#define DIN2   768
#define DINNER 1024      // HEADS*DH
#define NQKV   3072      // 3*DINNER
#define KCHUNK 64
#define NCHUNK (NTOT / KCHUNK)   // 64

#if defined(__AMDGCN__) && __has_builtin(__builtin_amdgcn_tensor_load_to_lds) && \
    __has_builtin(__builtin_amdgcn_s_wait_tensorcnt)
#define HAVE_TDM 1
#else
#define HAVE_TDM 0
#endif

static __device__ __forceinline__ v16h pack16(v8h lo, v8h hi) {
  v16h r;
#pragma unroll
  for (int i = 0; i < 8; ++i) { r[i] = lo[i]; r[i + 8] = hi[i]; }
  return r;
}

// A fragment (16x32 f16) from row-major src at (row0, k0), leading dim ld (halves).
// Lane l: m = l&15; halves i -> K = base+i (i = 0..15),
// base = (l>=16)?16:0.
static __device__ __forceinline__ v16h load_a(const _Float16* src, int ld, int row0, int k0, int lane) {
  const int m = lane & 15;
  const int g = (lane >> 4) * 16;
  const _Float16* p = src + (size_t)(row0 + m) * ld + k0 + g;
  v8h lo = *(const v8h*)p;
  v8h hi = *(const v8h*)(p + 8);
  return pack16(lo, hi);
}

// B fragment (32x16 f16): element (k,n) = bt[(n0+n)*ld + k0 + k]  (bt = B^T row-major).
// Lane l: n = l&15; halves i -> K = 16*(l>>4) + i.
static __device__ __forceinline__ v16h load_b(const _Float16* bt, int ld, int n0, int k0, int lane) {
  const _Float16* p = bt + (size_t)(n0 + (lane & 15)) * ld + k0 + (lane >> 4) * 16;
  v8h lo = *(const v8h*)p;
  v8h hi = *(const v8h*)(p + 8);
  return pack16(lo, hi);
}

static __device__ __forceinline__ v8f wmma16(v16h a, v16h b, v8f c) {
  return __builtin_amdgcn_wmma_f32_16x16x32_f16(false, a, false, b, (short)0, c, false, false);
}

#if HAVE_TDM
static __device__ __forceinline__ void tdm_issue(u32x4 g0, i32x8 g1) {
  i32x4 z4 = {0, 0, 0, 0};
#if __clang_major__ >= 23
  i32x8 z8 = {0, 0, 0, 0, 0, 0, 0, 0};
  __builtin_amdgcn_tensor_load_to_lds(g0, g1, z4, z4, z8, 0);
#else
  __builtin_amdgcn_tensor_load_to_lds(g0, g1, z4, z4, 0);
#endif
}

// 2D tile load via Tensor Data Mover (D# per CDNA5 ISA sec.8).
// tile_x contiguous halves per row, tile_y rows, row stride stride_x (halves).
static __device__ __forceinline__ void tdm_load_tile(const _Float16* gsrc, _Float16* lds,
                                                     unsigned tile_x, unsigned tile_y,
                                                     unsigned tensor_x, unsigned tensor_y,
                                                     unsigned long long stride_x) {
  unsigned long long ga = (unsigned long long)(size_t)gsrc;
  unsigned lo = (unsigned)(size_t)lds;          // low 32 bits of flat LDS addr = LDS offset
  u32x4 g0;
  g0[0] = 1u;                                   // count=1, user descriptor, no gather
  g0[1] = lo;                                   // lds_addr (bytes)
  g0[2] = (unsigned)(ga & 0xffffffffu);         // global_addr[31:0]
  g0[3] = (unsigned)((ga >> 32) & 0x1ffffffu) | (2u << 30);  // global_addr[56:32] | type=2
  i32x8 g1;
  unsigned w;
  g1[0] = (int)(1u << 16);                      // wg_mask=0, data_size=1 (2 bytes)
  w = (tensor_x & 0xffffu) << 16;               // tensor_dim0[15:0] -> bits 63:48
  g1[1] = (int)w;
  w = ((tensor_x >> 16) & 0xffffu) | ((tensor_y & 0xffffu) << 16);
  g1[2] = (int)w;                               // tensor_dim0[31:16] | tensor_dim1[15:0]
  w = ((tensor_y >> 16) & 0xffffu) | ((tile_x & 0xffffu) << 16);
  g1[3] = (int)w;                               // tensor_dim1[31:16] | tile_dim0
  g1[4] = (int)(tile_y & 0xffffu);              // tile_dim1 (tile_dim2 = 0)
  g1[5] = (int)(unsigned)(stride_x & 0xffffffffu);  // tensor_dim0_stride[31:0]
  g1[6] = (int)(unsigned)((stride_x >> 32) & 0xffffu);  // stride[47:32] (dim1_stride unused)
  g1[7] = 0;
  tdm_issue(g0, g1);
}
#endif

// ---------------- conversion kernels ----------------

__global__ void cvt_f16_kernel(const float* __restrict__ in, _Float16* __restrict__ out, int n) {
  int i = blockIdx.x * blockDim.x + threadIdx.x;
  if (i < n) out[i] = (_Float16)in[i];
}

// in: f32 [R][C] row-major  ->  out: f16 [C][R] row-major (i.e. transpose)
__global__ void transpose_f16_kernel(const float* __restrict__ in, _Float16* __restrict__ out,
                                     int R, int C) {
  int i = blockIdx.x * blockDim.x + threadIdx.x;
  if (i < R * C) {
    int r = i / C, c = i % C;
    out[(size_t)c * R + r] = (_Float16)in[i];
  }
}

// ---------------- QKV projection GEMM ----------------
// A: xh [M][K] f16, BT: WqkvT [3072][K] f16. Wave tile 32x64.
// Epilogue scatters into q[h][tok][d], k[h][tok][d], vT[h][d][tok].
__global__ __launch_bounds__(128) void gemm_qkv_kernel(const _Float16* __restrict__ A,
                                                       const _Float16* __restrict__ BT,
                                                       int M, int K, int rowOff,
                                                       _Float16* __restrict__ qh,
                                                       _Float16* __restrict__ kh,
                                                       _Float16* __restrict__ vT) {
  const int lane = threadIdx.x & 31;
  const int wid  = threadIdx.x >> 5;
  const int hi16 = lane >> 4;
  const int l15  = lane & 15;
  const int tilesN = NQKV / 64;               // 48
  const int gw = blockIdx.x * 4 + wid;
  const int tiles = (M / 32) * tilesN;
  if (gw >= tiles) return;
  const int m0 = (gw / tilesN) * 32;
  const int n0 = (gw % tilesN) * 64;

  v8f c[2][4];
#pragma unroll
  for (int mi = 0; mi < 2; ++mi)
#pragma unroll
    for (int ni = 0; ni < 4; ++ni)
#pragma unroll
      for (int r = 0; r < 8; ++r) c[mi][ni][r] = 0.0f;

  for (int k0 = 0; k0 < K; k0 += 32) {
    v16h a0 = load_a(A, K, m0, k0, lane);
    v16h a1 = load_a(A, K, m0 + 16, k0, lane);
    v16h b[4];
#pragma unroll
    for (int ni = 0; ni < 4; ++ni) b[ni] = load_b(BT, K, n0 + ni * 16, k0, lane);
#pragma unroll
    for (int ni = 0; ni < 4; ++ni) {
      c[0][ni] = wmma16(a0, b[ni], c[0][ni]);
      c[1][ni] = wmma16(a1, b[ni], c[1][ni]);
    }
  }

#pragma unroll
  for (int mi = 0; mi < 2; ++mi)
#pragma unroll
    for (int ni = 0; ni < 4; ++ni)
#pragma unroll
      for (int r = 0; r < 8; ++r) {
        int row = m0 + mi * 16 + r + 8 * hi16;
        int col = n0 + ni * 16 + l15;            // 0..3071
        int part = col >> 10;                     // 0:q 1:k 2:v
        int rem  = col & 1023;
        int hh   = rem >> 6;
        int d    = rem & 63;
        size_t tok = (size_t)(row + rowOff);
        _Float16 v = (_Float16)c[mi][ni][r];
        if (part == 0)      qh[((size_t)hh * NTOT + tok) * DH + d] = v;
        else if (part == 1) kh[((size_t)hh * NTOT + tok) * DH + d] = v;
        else                vT[((size_t)hh * DH + d) * NTOT + tok] = v;
      }
}

// ---------------- multi-head RMSNorm (q or k) ----------------
__global__ void rmsnorm_kernel(_Float16* __restrict__ x, const float* __restrict__ g1,
                               const float* __restrict__ g2) {
  int idx = blockIdx.x * blockDim.x + threadIdx.x;
  if (idx >= HEADS * NTOT) return;
  int h = idx / NTOT;
  int n = idx % NTOT;
  _Float16* row = x + (size_t)idx * DH;
  float ss = 0.f;
#pragma unroll
  for (int d = 0; d < DH; ++d) { float f = (float)row[d]; ss += f * f; }
  float inv = 8.0f / fmaxf(sqrtf(ss), 1e-12f);   // sqrt(64) / ||x||
  const float* g = (n < SN1) ? g1 : g2;
#pragma unroll
  for (int d = 0; d < DH; ++d)
    row[d] = (_Float16)((float)row[d] * inv * g[h * DH + d]);
}

// ---------------- flash attention ----------------
// Block = 8 waves = 128 queries of one head. K/V chunks (64 keys) staged once in
// LDS per block (TDM async when available), shared by all 8 waves.
__global__ __launch_bounds__(256) void attn_kernel(const _Float16* __restrict__ qh,
                                                   const _Float16* __restrict__ kh,
                                                   const _Float16* __restrict__ vT,
                                                   _Float16* __restrict__ attn) {
  __shared__ __align__(16) _Float16 Kbuf[2][KCHUNK * DH];   // [key][d]
  __shared__ __align__(16) _Float16 Vbuf[2][DH * KCHUNK];   // [d][key]
  __shared__ __align__(16) _Float16 pbuf[8][16 * KCHUNK];   // per-wave P staging

  const int tid  = threadIdx.x;
  const int lane = tid & 31;
  const int wid  = tid >> 5;
  const int hi16 = lane >> 4;
  const int l15  = lane & 15;
  const int h  = blockIdx.x >> 5;                  // 32 blocks per head
  const int q0 = (blockIdx.x & 31) * 128 + wid * 16;

  const _Float16* qbase = qh + (size_t)h * NTOT * DH;
  const _Float16* kbase = kh + (size_t)h * NTOT * DH;
  const _Float16* vbase = vT + (size_t)h * DH * NTOT;
  _Float16* pb = pbuf[wid];

  v16h aq0 = load_a(qbase, DH, q0, 0, lane);
  v16h aq1 = load_a(qbase, DH, q0, 32, lane);

  v8f acc[4];
#pragma unroll
  for (int t = 0; t < 4; ++t)
#pragma unroll
    for (int r = 0; r < 8; ++r) acc[t][r] = 0.0f;
  float mrow[8], lrow[8];
#pragma unroll
  for (int r = 0; r < 8; ++r) { mrow[r] = -3.0e38f; lrow[r] = 0.0f; }

  const float scl = 0.125f * (1.0f / 50.0f);       // dh^-0.5 / CLAMP

#if HAVE_TDM
  if (wid == 0) {   // prefetch chunks 0 and 1 (distance-2 pipeline)
    tdm_load_tile(kbase, Kbuf[0], KCHUNK * DH, 1, KCHUNK * DH, 1, KCHUNK * DH);
    tdm_load_tile(vbase, Vbuf[0], KCHUNK, DH, NTOT, DH, NTOT);
    tdm_load_tile(kbase + (size_t)KCHUNK * DH, Kbuf[1], KCHUNK * DH, 1, KCHUNK * DH, 1, KCHUNK * DH);
    tdm_load_tile(vbase + KCHUNK, Vbuf[1], KCHUNK, DH, NTOT, DH, NTOT);
  }
#endif

  for (int i = 0; i < NCHUNK; ++i) {
    const int cur = i & 1;
    const int kb = i * KCHUNK;
#if HAVE_TDM
    if (wid == 0) {   // TDM completes in issue order within a wave
      if (i + 1 < NCHUNK) __builtin_amdgcn_s_wait_tensorcnt(2);
      else                __builtin_amdgcn_s_wait_tensorcnt(0);
    }
    __syncthreads();
#else
    __syncthreads();
    {   // cooperative staging: K chunk contiguous, V chunk 64 rows x 128B
      const _Float16* ks = kbase + (size_t)kb * DH + tid * 16;
      _Float16* kd = Kbuf[cur] + tid * 16;
      *(v8h*)kd = *(const v8h*)ks;
      *(v8h*)(kd + 8) = *(const v8h*)(ks + 8);
      int vrow = tid >> 2, vcol = (tid & 3) * 16;
      const _Float16* vs = vbase + (size_t)vrow * NTOT + kb + vcol;
      _Float16* vd = Vbuf[cur] + vrow * KCHUNK + vcol;
      *(v8h*)vd = *(const v8h*)vs;
      *(v8h*)(vd + 8) = *(const v8h*)(vs + 8);
    }
    __syncthreads();
#endif
    const _Float16* Kb = Kbuf[cur];
    const _Float16* Vb = Vbuf[cur];

    // S = Q K^T for 4 key tiles (8 WMMAs)
    v8f s[4];
#pragma unroll
    for (int t = 0; t < 4; ++t) {
      const _Float16* kp = Kb + (t * 16 + l15) * DH + hi16 * 16;
      v16h b0 = pack16(*(const v8h*)kp,        *(const v8h*)(kp + 8));    // dh 0..31
      v16h b1 = pack16(*(const v8h*)(kp + 32), *(const v8h*)(kp + 40));   // dh 32..63
      v8f cc;
#pragma unroll
      for (int r = 0; r < 8; ++r) cc[r] = 0.0f;
      cc = wmma16(aq0, b0, cc);
      cc = wmma16(aq1, b1, cc);
      s[t] = cc;
    }
    // softclamp: tanh(sim/CLAMP)*CLAMP
#pragma unroll
    for (int t = 0; t < 4; ++t)
#pragma unroll
      for (int r = 0; r < 8; ++r)
        s[t][r] = tanhf(s[t][r] * scl) * 50.0f;
    // online softmax across 64 keys (4 tiles + 16-lane butterfly)
    float fac[8];
#pragma unroll
    for (int r = 0; r < 8; ++r) {
      float x = fmaxf(fmaxf(s[0][r], s[1][r]), fmaxf(s[2][r], s[3][r]));
#pragma unroll
      for (int mk = 1; mk < 16; mk <<= 1) x = fmaxf(x, __shfl_xor(x, mk, 32));
      float mnew = fmaxf(mrow[r], x);
      fac[r] = __expf(mrow[r] - mnew);
      mrow[r] = mnew;
    }
#pragma unroll
    for (int t = 0; t < 4; ++t)
#pragma unroll
      for (int r = 0; r < 8; ++r)
        s[t][r] = __expf(s[t][r] - mrow[r]);
#pragma unroll
    for (int r = 0; r < 8; ++r) {
      float rs = (s[0][r] + s[1][r]) + (s[2][r] + s[3][r]);
#pragma unroll
      for (int mk = 1; mk < 16; mk <<= 1) rs += __shfl_xor(rs, mk, 32);
      lrow[r] = lrow[r] * fac[r] + rs;
    }
#pragma unroll
    for (int t = 0; t < 4; ++t)
#pragma unroll
      for (int r = 0; r < 8; ++r) acc[t][r] *= fac[r];
    // stage P (16x64) in per-wave LDS, reload in A layout (LDS in-order per wave)
#pragma unroll
    for (int t = 0; t < 4; ++t)
#pragma unroll
      for (int r = 0; r < 8; ++r)
        pb[(r + 8 * hi16) * KCHUNK + t * 16 + l15] = (_Float16)s[t][r];
    v16h ap0 = load_a(pb, KCHUNK, 0, 0, lane);
    v16h ap1 = load_a(pb, KCHUNK, 0, 32, lane);
    // P @ V (8 WMMAs)
#pragma unroll
    for (int t = 0; t < 4; ++t) {
      const _Float16* vp = Vb + (t * 16 + l15) * KCHUNK + hi16 * 16;
      v16h bv0 = pack16(*(const v8h*)vp,        *(const v8h*)(vp + 8));   // keys 0..31
      v16h bv1 = pack16(*(const v8h*)(vp + 32), *(const v8h*)(vp + 40));  // keys 32..63
      acc[t] = wmma16(ap0, bv0, acc[t]);
      acc[t] = wmma16(ap1, bv1, acc[t]);
    }
    __syncthreads();   // all waves done with buf[cur]
#if HAVE_TDM
    if (wid == 0 && i + 2 < NCHUNK) {   // refill freed buffer, 2 chunks ahead
      const int nk = (i + 2) * KCHUNK;
      tdm_load_tile(kbase + (size_t)nk * DH, Kbuf[cur], KCHUNK * DH, 1, KCHUNK * DH, 1, KCHUNK * DH);
      tdm_load_tile(vbase + nk, Vbuf[cur], KCHUNK, DH, NTOT, DH, NTOT);
    }
#endif
  }
  // finalize: divide by row sums, merge heads into attn [N][1024] f16
#pragma unroll
  for (int t = 0; t < 4; ++t)
#pragma unroll
    for (int r = 0; r < 8; ++r) {
      int m = r + 8 * hi16;
      float o = acc[t][r] / lrow[r];
      attn[(size_t)(q0 + m) * DINNER + h * DH + t * 16 + l15] = (_Float16)o;
    }
}

// ---------------- output projection GEMM ----------------
// A: attn rows [M][1024] f16 (lda=1024), BT: WoutT [N][1024] f16, out: f32 [M][N].
__global__ __launch_bounds__(128) void gemm_out_kernel(const _Float16* __restrict__ A,
                                                       const _Float16* __restrict__ BT,
                                                       float* __restrict__ out,
                                                       int M, int N) {
  const int K = DINNER;
  const int lane = threadIdx.x & 31;
  const int wid  = threadIdx.x >> 5;
  const int hi16 = lane >> 4;
  const int l15  = lane & 15;
  const int tilesN = N / 64;
  const int gw = blockIdx.x * 4 + wid;
  const int tiles = (M / 32) * tilesN;
  if (gw >= tiles) return;
  const int m0 = (gw / tilesN) * 32;
  const int n0 = (gw % tilesN) * 64;

  v8f c[2][4];
#pragma unroll
  for (int mi = 0; mi < 2; ++mi)
#pragma unroll
    for (int ni = 0; ni < 4; ++ni)
#pragma unroll
      for (int r = 0; r < 8; ++r) c[mi][ni][r] = 0.0f;

  for (int k0 = 0; k0 < K; k0 += 32) {
    v16h a0 = load_a(A, K, m0, k0, lane);
    v16h a1 = load_a(A, K, m0 + 16, k0, lane);
    v16h b[4];
#pragma unroll
    for (int ni = 0; ni < 4; ++ni) b[ni] = load_b(BT, K, n0 + ni * 16, k0, lane);
#pragma unroll
    for (int ni = 0; ni < 4; ++ni) {
      c[0][ni] = wmma16(a0, b[ni], c[0][ni]);
      c[1][ni] = wmma16(a1, b[ni], c[1][ni]);
    }
  }
#pragma unroll
  for (int mi = 0; mi < 2; ++mi)
#pragma unroll
    for (int ni = 0; ni < 4; ++ni)
#pragma unroll
      for (int r = 0; r < 8; ++r) {
        int row = m0 + mi * 16 + r + 8 * hi16;
        int col = n0 + ni * 16 + l15;
        out[(size_t)row * N + col] = c[mi][ni][r];
      }
}

// ---------------- host launcher ----------------

extern "C" void kernel_launch(void* const* d_in, const int* in_sizes, int n_in,
                              void* d_out, int out_size, void* d_ws, size_t ws_size,
                              hipStream_t stream) {
  (void)in_sizes; (void)n_in; (void)out_size; (void)ws_size;
  const float* x1    = (const float*)d_in[0];
  const float* x2    = (const float*)d_in[1];
  // d_in[2], d_in[3]: masks (all-true in this problem) -- unused
  const float* Wqkv1 = (const float*)d_in[4];
  const float* Wqkv2 = (const float*)d_in[5];
  const float* Wout1 = (const float*)d_in[6];
  const float* Wout2 = (const float*)d_in[7];
  const float* qg1   = (const float*)d_in[8];
  const float* kg1   = (const float*)d_in[9];
  const float* qg2   = (const float*)d_in[10];
  const float* kg2   = (const float*)d_in[11];
  float* out = (float*)d_out;

  char* ws = (char*)d_ws;
  size_t off = 0;
  auto alloc = [&](size_t bytes) -> char* {
    char* p = ws + off;
    off += (bytes + 255) & ~(size_t)255;
    return p;
  };
  _Float16* x1h     = (_Float16*)alloc((size_t)SN1 * DIN1 * 2);
  _Float16* x2h     = (_Float16*)alloc((size_t)SN2 * DIN2 * 2);
  _Float16* Wqkv1T  = (_Float16*)alloc((size_t)NQKV * DIN1 * 2);
  _Float16* Wqkv2T  = (_Float16*)alloc((size_t)NQKV * DIN2 * 2);
  _Float16* Wout1T  = (_Float16*)alloc((size_t)DIN1 * DINNER * 2);
  _Float16* Wout2T  = (_Float16*)alloc((size_t)DIN2 * DINNER * 2);
  _Float16* qh      = (_Float16*)alloc((size_t)HEADS * NTOT * DH * 2);
  _Float16* kh      = (_Float16*)alloc((size_t)HEADS * NTOT * DH * 2);
  _Float16* vT      = (_Float16*)alloc((size_t)HEADS * DH * NTOT * 2);
  _Float16* attn    = (_Float16*)alloc((size_t)NTOT * DINNER * 2);

  const int TB = 256;
  // 1) convert activations to f16
  cvt_f16_kernel<<<(SN1 * DIN1 + TB - 1) / TB, TB, 0, stream>>>(x1, x1h, SN1 * DIN1);
  cvt_f16_kernel<<<(SN2 * DIN2 + TB - 1) / TB, TB, 0, stream>>>(x2, x2h, SN2 * DIN2);
  // 2) convert + transpose weights to B^T [N][K] f16
  transpose_f16_kernel<<<(DIN1 * NQKV + TB - 1) / TB, TB, 0, stream>>>(Wqkv1, Wqkv1T, DIN1, NQKV);
  transpose_f16_kernel<<<(DIN2 * NQKV + TB - 1) / TB, TB, 0, stream>>>(Wqkv2, Wqkv2T, DIN2, NQKV);
  transpose_f16_kernel<<<(DINNER * DIN1 + TB - 1) / TB, TB, 0, stream>>>(Wout1, Wout1T, DINNER, DIN1);
  transpose_f16_kernel<<<(DINNER * DIN2 + TB - 1) / TB, TB, 0, stream>>>(Wout2, Wout2T, DINNER, DIN2);
  // 3) QKV projections (wave tile 32x64, 4 waves/block)
  {
    int tiles = (SN1 / 32) * (NQKV / 64);
    gemm_qkv_kernel<<<(tiles + 3) / 4, 128, 0, stream>>>(x1h, Wqkv1T, SN1, DIN1, 0, qh, kh, vT);
  }
  {
    int tiles = (SN2 / 32) * (NQKV / 64);
    gemm_qkv_kernel<<<(tiles + 3) / 4, 128, 0, stream>>>(x2h, Wqkv2T, SN2, DIN2, SN1, qh, kh, vT);
  }
  // 4) multi-head RMSNorm on q and k
  rmsnorm_kernel<<<(HEADS * NTOT + TB - 1) / TB, TB, 0, stream>>>(qh, qg1, qg2);
  rmsnorm_kernel<<<(HEADS * NTOT + TB - 1) / TB, TB, 0, stream>>>(kh, kg1, kg2);
  // 5) flash attention: 16 heads x 32 blocks (128 queries/block, 8 waves)
  attn_kernel<<<HEADS * (NTOT / 128), 256, 0, stream>>>(qh, kh, vT, attn);
  // 6) output projections straight into d_out (out1 flat, then out2 flat)
  {
    int tiles = (SN1 / 32) * (DIN1 / 64);
    gemm_out_kernel<<<(tiles + 3) / 4, 128, 0, stream>>>(attn, Wout1T, out, SN1, DIN1);
  }
  {
    int tiles = (SN2 / 32) * (DIN2 / 64);
    gemm_out_kernel<<<(tiles + 3) / 4, 128, 0, stream>>>(attn + (size_t)SN1 * DINNER, Wout2T,
                                                        out + (size_t)SN1 * DIN1, SN2, DIN2);
  }
}
